// NAGphormer_71786083385393
// MI455X (gfx1250) — compile-verified, matrix operand order checked by reading, not run on hardware
//
#include <hip/hip_runtime.h>
#include <hip/hip_bf16.h>

typedef __attribute__((ext_vector_type(16))) __bf16 v16bf;
typedef __attribute__((ext_vector_type(8)))  float  v8f;

#define BS_ROW   80                       // bytes per B row in LDS image (32 bf16 = 64B + 16B pad)
#define WCHUNK   (256 * BS_ROW)           // 20480 B per K-chunk image
#define WSTAGE_BYTES (16 * WCHUNK)        // 327680 B total weight image

// ---------------------------------------------------------------------------
// helpers
// ---------------------------------------------------------------------------
__device__ __forceinline__ float wave_sum32(float v) {
#pragma unroll
  for (int off = 16; off > 0; off >>= 1)
    v += __shfl_xor(v, off, 32);
  return v;
}

// pack two f32 -> two bf16 (round-to-nearest-even) into one u32 (lo = a)
__device__ __forceinline__ unsigned f2bf2(float a, float b) {
  unsigned ua = __float_as_uint(a);
  unsigned ub = __float_as_uint(b);
  ua = (ua + 0x7FFFu + ((ua >> 16) & 1u)) >> 16;
  ub = (ub + 0x7FFFu + ((ub >> 16) & 1u)) >> 16;
  return (ua & 0xFFFFu) | (ub << 16);
}

// LDS byte offset of a pointer into a __shared__ array
__device__ __forceinline__ unsigned lds_offset(const void* p) {
  return (unsigned)(size_t)(__attribute__((address_space(3))) const void*)p;
}

// ---------------------------------------------------------------------------
// Kernel 1: fused LayerNorm + attention softmax pooling.
// One wave32 per node. Lane holds, for each of the 8 rows, 16 floats at
// h = c*128 + lane*4 + i  (c=0..3, i=0..3).  Writes fused row as bf16.
// ---------------------------------------------------------------------------
__global__ __launch_bounds__(256)
void nag_prep_kernel(const float* __restrict__ x,
                     const float* __restrict__ gamma,
                     const float* __restrict__ beta,
                     const float* __restrict__ attn_W,   // [1024] : W_t=[0:512), W_n=[512:1024)
                     const float* __restrict__ attn_b,   // [1]
                     uint2* __restrict__ fused_out)      // bf16 [N][512] viewed as uint2
{
  const int lane = threadIdx.x & 31;
  const int wid  = threadIdx.x >> 5;
  const size_t node = (size_t)blockIdx.x * 8 + wid;

  float v[8][16];
  float mean[8], inv[8];

  // ---- pass 1: load + mean/var per row -------------------------------
  const float4* xp = (const float4*)x + node * 1024;   // 8*512/4 float4 per node
#pragma unroll
  for (int r = 0; r < 8; ++r) {
    float s = 0.f, sq = 0.f;
#pragma unroll
    for (int c = 0; c < 4; ++c) {
      float4 f = xp[(size_t)r * 128 + c * 32 + lane];
      v[r][c * 4 + 0] = f.x; v[r][c * 4 + 1] = f.y;
      v[r][c * 4 + 2] = f.z; v[r][c * 4 + 3] = f.w;
      s  += f.x + f.y + f.z + f.w;
      sq += f.x * f.x + f.y * f.y + f.z * f.z + f.w * f.w;
    }
    s  = wave_sum32(s);
    sq = wave_sum32(sq);
    const float m   = s * (1.f / 512.f);
    const float var = sq * (1.f / 512.f) - m * m;
    mean[r] = m;
    inv[r]  = rsqrtf(var + 1e-5f);
  }

  // ---- pass 2: normalize in-place + attention dots -------------------
  float dotv[8];
#pragma unroll
  for (int r = 0; r < 8; ++r) dotv[r] = 0.f;

#pragma unroll
  for (int c = 0; c < 4; ++c) {
    float4 g4  = ((const float4*)gamma)[c * 32 + lane];
    float4 b4  = ((const float4*)beta )[c * 32 + lane];
    float4 wt4 = ((const float4*)attn_W)[c * 32 + lane];          // W_t
    float4 wn4 = ((const float4*)(attn_W + 512))[c * 32 + lane];  // W_n
    const float ga[4] = {g4.x, g4.y, g4.z, g4.w};
    const float be[4] = {b4.x, b4.y, b4.z, b4.w};
    const float wt[4] = {wt4.x, wt4.y, wt4.z, wt4.w};
    const float wn[4] = {wn4.x, wn4.y, wn4.z, wn4.w};
#pragma unroll
    for (int r = 0; r < 8; ++r) {
#pragma unroll
      for (int i = 0; i < 4; ++i) {
        float nv = (v[r][c * 4 + i] - mean[r]) * inv[r] * ga[i] + be[i];
        v[r][c * 4 + i] = nv;
        dotv[r] += nv * ((r == 0) ? wt[i] : wn[i]);
      }
    }
  }

  float dot[8];
#pragma unroll
  for (int r = 0; r < 8; ++r) dot[r] = wave_sum32(dotv[r]);

  // ---- softmax over 7 neighbors (uniform across lanes) ---------------
  const float ab = attn_b[0];
  const float t  = dot[0];
  float lg[7];
  float mx = -3.402823466e38f;
#pragma unroll
  for (int j = 0; j < 7; ++j) {
    lg[j] = dot[j + 1] + t + ab;
    mx = fmaxf(mx, lg[j]);
  }
  float se = 0.f;
#pragma unroll
  for (int j = 0; j < 7; ++j) {
    lg[j] = __expf(lg[j] - mx);
    se += lg[j];
  }
  const float rs = 1.f / se;

  // ---- fused = node + sum_j a_j * neighbor_j ; store as bf16 ---------
#pragma unroll
  for (int c = 0; c < 4; ++c) {
    float f[4];
#pragma unroll
    for (int i = 0; i < 4; ++i) {
      float acc = v[0][c * 4 + i];
#pragma unroll
      for (int j = 0; j < 7; ++j)
        acc += (lg[j] * rs) * v[j + 1][c * 4 + i];
      f[i] = acc;
    }
    uint2 pkt;
    pkt.x = f2bf2(f[0], f[1]);
    pkt.y = f2bf2(f[2], f[3]);
    fused_out[node * 128 + c * 32 + lane] = pkt;   // 8B per lane, coalesced
  }
}

// ---------------------------------------------------------------------------
// Kernel 2: one-time weight staging.  Converts out_W (f32 [512][256]) into
// the exact padded LDS image the GEMM wants:
//   wstage[kc][n][kk]  (bf16, rows padded to 80B), kc=0..15, n=0..255, kk=0..31
// Block kc (16 blocks), thread n (256 threads).
// ---------------------------------------------------------------------------
__global__ __launch_bounds__(256)
void nag_wstage_kernel(const float* __restrict__ out_W,
                       unsigned char* __restrict__ wstage) {
  const int kc = blockIdx.x;     // 0..15
  const int n  = threadIdx.x;    // 0..255
  unsigned char* row = wstage + (size_t)kc * WCHUNK + (size_t)n * BS_ROW;
#pragma unroll
  for (int i = 0; i < 32; i += 2) {
    const float f0 = out_W[(size_t)(kc * 32 + i    ) * 256 + n];
    const float f1 = out_W[(size_t)(kc * 32 + i + 1) * 256 + n];
    *(unsigned*)(row + i * 2) = f2bf2(f0, f1);
  }
}

// ---------------------------------------------------------------------------
// Kernel 3: GEMM  out = relu(fused(bf16)[65536x512] @ out_W[512x256] + out_b)
// using v_wmma_f32_16x16x32_bf16.  8 waves/block; wave w owns M-tile of 16
// rows and all 256 columns (16 accumulators).  Per K-chunk the pre-staged
// 20KB weight image is copied global->LDS with async-LDS loads (ASYNCcnt).
// ---------------------------------------------------------------------------
__global__ __launch_bounds__(256)
void nag_gemm_kernel(const unsigned short* __restrict__ Abf,  // bf16 [M][512]
                     const unsigned char* __restrict__ wstage,// padded bf16 image
                     const float* __restrict__ out_b,         // f32  [256]
                     float* __restrict__ out)                 // f32  [M][256]
{
  __shared__ __align__(16) unsigned char Bs[256 * BS_ROW];    // 20 KB

  const int tid  = threadIdx.x;
  const int lane = tid & 31;
  const int wid  = tid >> 5;
  const int half = lane >> 4;     // 0: lanes 0-15, 1: lanes 16-31
  const int l16  = lane & 15;
  const size_t m0 = (size_t)blockIdx.x * 128 + (size_t)wid * 16;

  const unsigned bs_base = lds_offset(Bs);

  v8f acc[16] = {};

  for (int kc = 0; kc < 16; ++kc) {
    __syncthreads();   // protect Bs reads from previous iteration

    // ---- async copy of 20KB weight chunk: global -> LDS --------------
    {
      const unsigned long long gbase =
          (unsigned long long)(size_t)(wstage + (size_t)kc * WCHUNK);
#pragma unroll
      for (int it = 0; it < 5; ++it) {
        const unsigned off = (unsigned)(it * 4096 + tid * 16);  // 5*256*16B = 20480B
        const unsigned      laddr = bs_base + off;
        const unsigned long long gaddr = gbase + off;
        asm volatile("global_load_async_to_lds_b128 %0, %1, off"
                     :: "v"(laddr), "v"(gaddr)
                     : "memory");
      }
      asm volatile("s_wait_asynccnt 0x0" ::: "memory");
    }
    __syncthreads();

    // ---- A fragment: row m0+l16; K pairs (half*8..+7), (16+half*8..+7)
    union { uint4 q[2]; v16bf v; } Au;
    const unsigned short* ap = Abf + (m0 + (size_t)l16) * 512 + kc * 32 + half * 8;
    Au.q[0] = *(const uint4*)(ap);        // K = kc*32 + half*8 .. +7
    Au.q[1] = *(const uint4*)(ap + 16);   // K = kc*32 + 16 + half*8 .. +7

#pragma unroll
    for (int nt = 0; nt < 16; ++nt) {
      const int n = nt * 16 + l16;
      union { uint4 q[2]; v16bf v; } Bu;
      Bu.q[0] = *(const uint4*)(&Bs[n * BS_ROW + half * 16]);       // kk = half*8..+7
      Bu.q[1] = *(const uint4*)(&Bs[n * BS_ROW + 32 + half * 16]);  // kk = 16+half*8..+7
      acc[nt] = __builtin_amdgcn_wmma_f32_16x16x32_bf16(
          /*neg_a=*/false, Au.v, /*neg_b=*/false, Bu.v,
          /*c_mod=*/(short)0, acc[nt], /*reuse_a=*/false, /*reuse_b=*/false);
    }
  }

  // ---- epilogue: VGPR r holds row (half? 8+r : r), col = nt*16 + l16 --
  const size_t rowbase = m0 + (half ? 8 : 0);
#pragma unroll
  for (int nt = 0; nt < 16; ++nt) {
    const int n = nt * 16 + l16;
    const float bias = out_b[n];
#pragma unroll
    for (int r = 0; r < 8; ++r) {
      float vv = acc[nt][r] + bias;
      vv = fmaxf(vv, 0.f);
      out[(rowbase + r) * 256 + n] = vv;
    }
  }
}

// ---------------------------------------------------------------------------
// launcher
// ---------------------------------------------------------------------------
extern "C" void kernel_launch(void* const* d_in, const int* in_sizes, int n_in,
                              void* d_out, int out_size, void* d_ws, size_t ws_size,
                              hipStream_t stream) {
  const float* x      = (const float*)d_in[0];
  const float* gamma  = (const float*)d_in[1];
  const float* beta   = (const float*)d_in[2];
  const float* attn_W = (const float*)d_in[3];
  const float* attn_b = (const float*)d_in[4];
  const float* out_W  = (const float*)d_in[5];
  const float* out_b  = (const float*)d_in[6];
  float* out = (float*)d_out;

  const int N = in_sizes[0] / (8 * 512);   // 65536

  // ws layout: [0, 320KB) weight image, then bf16 fused activations
  unsigned char*  wstage     = (unsigned char*)d_ws;
  unsigned short* fused_bf16 = (unsigned short*)((unsigned char*)d_ws + WSTAGE_BYTES);

  nag_wstage_kernel<<<16, 256, 0, stream>>>(out_W, wstage);
  nag_prep_kernel<<<N / 8, 256, 0, stream>>>(x, gamma, beta, attn_W, attn_b,
                                             (uint2*)fused_bf16);
  nag_gemm_kernel<<<N / 128, 256, 0, stream>>>(fused_bf16, wstage, out_b, out);
}
